// MIDIStatisticalFeatures_15152644621094
// MI455X (gfx1250) — compile-verified, matrix operand order checked by reading, not run on hardware
//
#include <hip/hip_runtime.h>
#include <hip/hip_bf16.h>
#include <math.h>

// ---------------------------------------------------------------------------
// Types for CDNA5 WMMA (wave32): v_wmma_f32_16x16x32_bf16
// ---------------------------------------------------------------------------
typedef __attribute__((ext_vector_type(16))) __bf16 v16bf;
typedef __attribute__((ext_vector_type(8)))  float  v8f;

union FragAB {
    unsigned int u[8];
    v16bf v;
};

static __device__ __forceinline__ unsigned short f2bf(float f) {
    // round-to-nearest-even float32 -> bf16
    unsigned u = __builtin_bit_cast(unsigned, f);
    unsigned r = (u + 0x7FFFu + ((u >> 16) & 1u)) >> 16;
    return (unsigned short)r;
}

// ---------------------------------------------------------------------------
// Threefry-2x32 (JAX default PRNG), 20 rounds, matching jax/_src/prng.py
// ---------------------------------------------------------------------------
static __device__ __forceinline__ void tf_round(unsigned &x0, unsigned &x1, int r) {
    x0 += x1;
    x1 = (x1 << r) | (x1 >> (32 - r));
    x1 ^= x0;
}

static __device__ __forceinline__ void threefry2x32(unsigned k0, unsigned k1,
                                                    unsigned c0, unsigned c1,
                                                    unsigned &o0, unsigned &o1) {
    unsigned k2 = k0 ^ k1 ^ 0x1BD11BDAu;
    unsigned x0 = c0 + k0, x1 = c1 + k1;
    tf_round(x0, x1, 13); tf_round(x0, x1, 15); tf_round(x0, x1, 26); tf_round(x0, x1, 6);
    x0 += k1; x1 += k2 + 1u;
    tf_round(x0, x1, 17); tf_round(x0, x1, 29); tf_round(x0, x1, 16); tf_round(x0, x1, 24);
    x0 += k2; x1 += k0 + 2u;
    tf_round(x0, x1, 13); tf_round(x0, x1, 15); tf_round(x0, x1, 26); tf_round(x0, x1, 6);
    x0 += k0; x1 += k1 + 3u;
    tf_round(x0, x1, 17); tf_round(x0, x1, 29); tf_round(x0, x1, 16); tf_round(x0, x1, 24);
    x0 += k1; x1 += k2 + 4u;
    tf_round(x0, x1, 13); tf_round(x0, x1, 15); tf_round(x0, x1, 26); tf_round(x0, x1, 6);
    x0 += k2; x1 += k0 + 5u;
    o0 = x0; o1 = x1;
}

static __device__ __forceinline__ float jax_uniform_bits(unsigned bits) {
    return __builtin_bit_cast(float, (bits >> 9) | 0x3f800000u) - 1.0f;
}

// jax.random.uniform(key(42), (1024,12)) flat element i of 12288
static __device__ __forceinline__ float harmony_value(unsigned i) {
    const unsigned HALF = 6144u;          // 1024*12/2
    unsigned c0, c1, o0, o1;
    if (i < HALF) { c0 = i;        c1 = i + HALF; }
    else          { c0 = i - HALF; c1 = i;        }
    threefry2x32(0u, 42u, c0, c1, o0, o1);
    return jax_uniform_bits(i < HALF ? o0 : o1);
}

// ---------------------------------------------------------------------------
// Constants
// ---------------------------------------------------------------------------
#define B_ROWS     1024
#define T_LEN      8192
#define K_PAD      160     // 150 stats dims padded to multiple of 32
#define HID        256
#define OUT_N      128

// ---------------------------------------------------------------------------
// Kernel 1: per-row statistics -> bf16 stats[B][160]
// ---------------------------------------------------------------------------
__global__ void stats_kernel(const int* __restrict__ tok,
                             unsigned short* __restrict__ statsBF) {
    __shared__ unsigned hist[128];
    __shared__ unsigned ssum, ssq;

    const int row = blockIdx.x;
    const int tid = threadIdx.x;

    if (tid < 128) hist[tid] = 0u;
    if (tid == 0) { ssum = 0u; ssq = 0u; }
    __syncthreads();

    const int* rtok = tok + (size_t)row * T_LEN;
    unsigned ls = 0u, lq = 0u;
#pragma unroll 4
    for (int j = 0; j < T_LEN / 256; ++j) {
        unsigned t = (unsigned)rtok[j * 256 + tid];   // tokens in [0, 390)
        atomicAdd(&hist[t & 127u], 1u);
        ls += t;
        lq += t * t;
    }
    atomicAdd(&ssum, ls);
    atomicAdd(&ssq, lq);
    __syncthreads();

    unsigned short* srow = statsBF + (size_t)row * K_PAD;
    const float inv = 1.0f / (8192.0f + 1e-8f);

    if (tid < 128) {
        srow[tid] = f2bf((float)hist[tid] * inv);           // pitch distribution
    } else if (tid == 128) {
        double s = (double)ssum;
        srow[128] = f2bf((float)(s / (double)T_LEN));       // rhythm mean
    } else if (tid == 129) {
        double s = (double)ssum, q = (double)ssq;
        double var = (q - s * s / (double)T_LEN) / (double)(T_LEN - 1);
        srow[129] = f2bf((float)sqrt(var));                 // rhythm std (unbiased)
    } else if (tid < 138) {
        srow[tid] = 0;                                      // rhythm zeros
    } else if (tid < 150) {
        unsigned j = (unsigned)(tid - 138);
        srow[tid] = f2bf(harmony_value((unsigned)row * 12u + j));  // harmony
    } else if (tid < K_PAD) {
        srow[tid] = 0;                                      // K padding
    }
}

// ---------------------------------------------------------------------------
// Kernel 2: convert weights to bf16 (W1 K-padded 150 -> 160 rows)
// ---------------------------------------------------------------------------
__global__ void convert_weights(const float* __restrict__ W1,
                                const float* __restrict__ W2,
                                unsigned short* __restrict__ W1bf,
                                unsigned short* __restrict__ W2bf) {
    const int idx = blockIdx.x * blockDim.x + threadIdx.x;
    const int W1N = K_PAD * HID;      // 40960
    const int W2N = HID * OUT_N;      // 32768
    if (idx < W1N) {
        int k = idx / HID;
        int n = idx - k * HID;
        W1bf[idx] = (k < 150) ? f2bf(W1[k * HID + n]) : (unsigned short)0;
    } else if (idx < W1N + W2N) {
        int i2 = idx - W1N;
        W2bf[i2] = f2bf(W2[i2]);
    }
}

// ---------------------------------------------------------------------------
// Kernel 3: fused MLP, WMMA bf16. One block = 16 output rows, 8 waves.
//   stage 1: h = relu(stats @ W1 + b1)   (16 x 256, bf16 in LDS)
//   stage 2: out = h @ W2 + b2           (16 x 128, f32)
// ---------------------------------------------------------------------------
__global__ void fused_mlp(const unsigned short* __restrict__ statsBF,
                          const unsigned short* __restrict__ W1bf,
                          const unsigned short* __restrict__ W2bf,
                          const float* __restrict__ b1,
                          const float* __restrict__ b2,
                          float* __restrict__ out) {
    __shared__ unsigned short a_lds[16 * K_PAD];   // 5 KB
    __shared__ unsigned short h_lds[16 * HID];     // 8 KB

    const int tid     = threadIdx.x;
    const int wave    = tid >> 5;
    const int lane    = tid & 31;
    const int lm      = lane & 15;      // row (A/C) or col (B/C) within tile
    const int kh      = lane >> 4;      // K-half selector / C row-half
    const int rowbase = blockIdx.x * 16;

    // ---- load 16x160 bf16 A tile (contiguous in global) into LDS ----
    {
        const unsigned* src = (const unsigned*)(statsBF + (size_t)rowbase * K_PAD);
        unsigned* dst = (unsigned*)a_lds;
#pragma unroll
        for (int i = 0; i < (16 * K_PAD / 2) / 256; ++i)
            dst[i * 256 + tid] = src[i * 256 + tid];
    }
    __syncthreads();

    // K offsets per A/B fragment register v (ISA 7.12.2, 16-bit, wave32):
    //   koff(v) = 2*(v&3) + 16*(v>>2) + 8*kh
    // ---- stage 1: each wave computes hidden cols [wave*32, wave*32+32) ----
    {
        v8f c0 = {}; v8f c1 = {};
        const int nb0 = wave * 32;
        const int nb1 = wave * 32 + 16;
#pragma unroll
        for (int kb = 0; kb < K_PAD; kb += 32) {
            FragAB a, bA, bB;
#pragma unroll
            for (int v = 0; v < 8; ++v) {
                int koff = 2 * (v & 3) + 16 * (v >> 2) + 8 * kh;
                int k = kb + koff;
                a.u[v] = *(const unsigned*)&a_lds[lm * K_PAD + k];
                unsigned lo0 = W1bf[(size_t)k * HID + nb0 + lm];
                unsigned hi0 = W1bf[(size_t)(k + 1) * HID + nb0 + lm];
                bA.u[v] = lo0 | (hi0 << 16);
                unsigned lo1 = W1bf[(size_t)k * HID + nb1 + lm];
                unsigned hi1 = W1bf[(size_t)(k + 1) * HID + nb1 + lm];
                bB.u[v] = lo1 | (hi1 << 16);
            }
            c0 = __builtin_amdgcn_wmma_f32_16x16x32_bf16(false, a.v, false, bA.v,
                                                         (short)0, c0, false, false);
            c1 = __builtin_amdgcn_wmma_f32_16x16x32_bf16(false, a.v, false, bB.v,
                                                         (short)0, c1, false, false);
        }
        // bias + relu, park bf16 hidden tile in LDS
        float bias0 = b1[nb0 + lm];
        float bias1 = b1[nb1 + lm];
#pragma unroll
        for (int r = 0; r < 8; ++r) {
            int m = r + kh * 8;
            float h0 = c0[r] + bias0; h0 = h0 > 0.0f ? h0 : 0.0f;
            float h1 = c1[r] + bias1; h1 = h1 > 0.0f ? h1 : 0.0f;
            h_lds[m * HID + nb0 + lm] = f2bf(h0);
            h_lds[m * HID + nb1 + lm] = f2bf(h1);
        }
    }
    __syncthreads();

    // ---- stage 2: each wave computes output cols [wave*16, wave*16+16) ----
    {
        v8f c = {};
        const int nb = wave * 16;
#pragma unroll
        for (int kb = 0; kb < HID; kb += 32) {
            FragAB a, b;
#pragma unroll
            for (int v = 0; v < 8; ++v) {
                int koff = 2 * (v & 3) + 16 * (v >> 2) + 8 * kh;
                int k = kb + koff;
                a.u[v] = *(const unsigned*)&h_lds[lm * HID + k];
                unsigned lo = W2bf[(size_t)k * OUT_N + nb + lm];
                unsigned hi = W2bf[(size_t)(k + 1) * OUT_N + nb + lm];
                b.u[v] = lo | (hi << 16);
            }
            c = __builtin_amdgcn_wmma_f32_16x16x32_bf16(false, a.v, false, b.v,
                                                        (short)0, c, false, false);
        }
        float bias = b2[nb + lm];
#pragma unroll
        for (int r = 0; r < 8; ++r) {
            int m = r + kh * 8;
            out[(size_t)(rowbase + m) * OUT_N + nb + lm] = c[r] + bias;
        }
    }
}

// ---------------------------------------------------------------------------
// Launch
// ---------------------------------------------------------------------------
extern "C" void kernel_launch(void* const* d_in, const int* in_sizes, int n_in,
                              void* d_out, int out_size, void* d_ws, size_t ws_size,
                              hipStream_t stream) {
    const int*   tok = (const int*)d_in[0];
    const float* W1  = (const float*)d_in[1];
    const float* b1  = (const float*)d_in[2];
    const float* W2  = (const float*)d_in[3];
    const float* b2  = (const float*)d_in[4];
    float* out = (float*)d_out;

    // workspace layout (256-byte aligned offsets)
    unsigned short* statsBF = (unsigned short*)d_ws;                          // 1024*160*2 = 327680 B
    unsigned short* W1bf    = (unsigned short*)((char*)d_ws + 327680);        // 160*256*2  =  81920 B
    unsigned short* W2bf    = (unsigned short*)((char*)d_ws + 409600);        // 256*128*2  =  65536 B

    stats_kernel<<<B_ROWS, 256, 0, stream>>>(tok, statsBF);
    convert_weights<<<(K_PAD * HID + HID * OUT_N + 255) / 256, 256, 0, stream>>>(W1, W2, W1bf, W2bf);
    fused_mlp<<<B_ROWS / 16, 256, 0, stream>>>(statsBF, W1bf, W2bf, b1, b2, out);
}
